// VideoChapterWindowAttention_63187558859183
// MI455X (gfx1250) — compile-verified, matrix operand order checked by reading, not run on hardware
//
#include <hip/hip_runtime.h>
#include <hip/hip_bf16.h>

// ---------------------------------------------------------------------------
// Types & helpers
// ---------------------------------------------------------------------------
typedef __bf16 bf16_t;
typedef __attribute__((ext_vector_type(16))) __bf16 v16bf;
typedef __attribute__((ext_vector_type(8)))  float  v8f;
typedef __attribute__((ext_vector_type(4)))  unsigned int u32x4;
typedef __attribute__((ext_vector_type(8)))  int i32x8;
typedef __attribute__((ext_vector_type(4)))  int i32x4;

static __device__ __forceinline__ bf16_t f2bf(float f) {
  union { float f; unsigned u; } x; x.f = f;
  unsigned r = x.u + 0x7FFFu + ((x.u >> 16) & 1u);   // round-to-nearest-even
  unsigned short h = (unsigned short)(r >> 16);
  return __builtin_bit_cast(bf16_t, h);
}
static __device__ __forceinline__ float bf2f(bf16_t b) {
  unsigned short h = __builtin_bit_cast(unsigned short, b);
  union { unsigned u; float f; } x; x.u = ((unsigned)h) << 16;
  return x.f;
}

// block-wide sum over 256 threads (red must be float[256] in LDS)
static __device__ __forceinline__ float block_reduce_sum(float v, float* red) {
  int t = threadIdx.x;
  __syncthreads();
  red[t] = v; __syncthreads();
#pragma unroll
  for (int k = 128; k > 0; k >>= 1) {
    if (t < k) red[t] += red[t + k];
    __syncthreads();
  }
  float r = red[0];
  __syncthreads();
  return r;
}

// ---------------------------------------------------------------------------
// Tensor Data Mover: DMA a 2D tile (tile_d1 rows x tile_d0 elems, bf16) from
// global [tensor_d1][tensor_d0] (row stride = tensor_d0) into LDS at lds_addr.
// D# layout per CDNA5 ISA 8.3/8.4 (group0: count/lds/global/type,
// group1: data_size, dims, tile dims, strides). Remaining groups zero (2D).
// This toolchain's builtin takes 6 args (g0,g1,g2,g3,extra,cpol).
// ---------------------------------------------------------------------------
static __device__ __forceinline__ void tdm_load_2d_bf16(
    unsigned lds_addr, const void* gaddr, unsigned tensor_d0,
    unsigned tensor_d1, unsigned tile_d0, unsigned tile_d1)
{
  unsigned long long ga = (unsigned long long)(size_t)gaddr;
  u32x4 g0;
  g0[0] = 1u;                                   // count=1, user mode
  g0[1] = lds_addr;                             // LDS byte address
  g0[2] = (unsigned)ga;                         // global_addr[31:0]
  g0[3] = (unsigned)(ga >> 32) | 0x80000000u;   // global_addr[56:32] | type=2
  i32x8 g1;
  g1[0] = (int)(1u << 16);                      // data_size=1 (2 bytes)
  g1[1] = (int)(tensor_d0 << 16);               // tensor_dim0[15:0]
  g1[2] = (int)((tensor_d0 >> 16) | (tensor_d1 << 16));  // d0 hi | d1 lo
  g1[3] = (int)((tensor_d1 >> 16) | (tile_d0 << 16));    // d1 hi | tile_dim0
  g1[4] = (int)tile_d1;                         // tile_dim1 (tile_dim2=0)
  g1[5] = (int)tensor_d0;                       // tensor_dim0_stride[31:0]
  g1[6] = 0;                                    // stride hi, dim1_stride lo
  g1[7] = 0;
  i32x4 z4 = {0, 0, 0, 0};
  i32x8 z8 = {0, 0, 0, 0, 0, 0, 0, 0};
  __builtin_amdgcn_tensor_load_to_lds(g0, g1, z4, z4, z8, 0);
}

// ---------------------------------------------------------------------------
// Constants for this problem
// ---------------------------------------------------------------------------
#define B_   4096
#define W_   33
#define H_   1024
#define NH_  16
#define DH_  64
#define MID_ 16

// ---------------------------------------------------------------------------
// Kernel 1: positional-encoding MLP (batch independent): pe[33][1024]
// ---------------------------------------------------------------------------
__global__ __launch_bounds__(256) void pe_kernel(
    const float* __restrict__ pe_w1, const float* __restrict__ pe_b1,
    const float* __restrict__ pe_g1, const float* __restrict__ pe_be1,
    const float* __restrict__ pe_w2, const float* __restrict__ pe_b2,
    const float* __restrict__ pe_g2, const float* __restrict__ pe_be2,
    float* __restrict__ pe_out)
{
  __shared__ float h1[512];
  __shared__ float h2[1024];
  __shared__ float red[256];
  int w = blockIdx.x, t = threadIdx.x;
  float local = ((float)w - 16.0f) / (16.0f + 1e-6f);

  for (int c = t; c < 512; c += 256) h1[c] = local * pe_w1[c] + pe_b1[c];
  __syncthreads();

  float s = 0.f, s2 = 0.f;
  for (int c = t; c < 512; c += 256) { float v = h1[c]; s += v; s2 += v * v; }
  float mean = block_reduce_sum(s, red) * (1.0f / 512.0f);
  float m2   = block_reduce_sum(s2, red) * (1.0f / 512.0f);
  float inv  = rsqrtf(m2 - mean * mean + 1e-5f);
  for (int c = t; c < 512; c += 256)
    h1[c] = tanhf((h1[c] - mean) * inv * pe_g1[c] + pe_be1[c]);
  __syncthreads();

  for (int c = t; c < 1024; c += 256) {
    float a = pe_b2[c];
    for (int k = 0; k < 512; ++k) a += h1[k] * pe_w2[k * 1024 + c];
    h2[c] = a;
  }
  __syncthreads();

  s = 0.f; s2 = 0.f;
  for (int c = t; c < 1024; c += 256) { float v = h2[c]; s += v; s2 += v * v; }
  mean = block_reduce_sum(s, red) * (1.0f / 1024.0f);
  m2   = block_reduce_sum(s2, red) * (1.0f / 1024.0f);
  inv  = rsqrtf(m2 - mean * mean + 1e-5f);
  for (int c = t; c < 1024; c += 256)
    pe_out[w * 1024 + c] = (h2[c] - mean) * inv * pe_g2[c] + pe_be2[c];
}

// ---------------------------------------------------------------------------
// Kernel 2: weight convert + transpose: Wt[n][k] = bf16(W[k][n])
// ---------------------------------------------------------------------------
__global__ __launch_bounds__(256) void conv_transpose_kernel(
    const float* __restrict__ Win, bf16_t* __restrict__ Wt, int K, int N)
{
  size_t idx = (size_t)blockIdx.x * 256 + threadIdx.x;
  size_t total = (size_t)K * N;
  if (idx >= total) return;
  size_t k = idx / (size_t)N;
  size_t n = idx - k * (size_t)N;
  Wt[n * (size_t)K + k] = f2bf(Win[idx]);
}

// ---------------------------------------------------------------------------
// Kernel 3: fused per-batch attention
//   x = hidden[b] + pe  (bf16 in LDS)
//   K = x @ wk + bk,  V = x @ wv + bv   via WMMA (bf16)
//   q = x[16] @ wq + bq                 (scalar, f32)
//   scores/softmax/context              (tiny, scalar)
// Dynamic LDS layout:
//   xs bf16[33*1024] | ks bf16[33*1024] | vs bf16[33*1024] | qf f32[1024] | sc f32[16*40]
// ---------------------------------------------------------------------------
__global__ __launch_bounds__(256) void attn_kernel(
    const float* __restrict__ hidden, const float* __restrict__ pe,
    const float* __restrict__ wq, const float* __restrict__ bq,
    const bf16_t* __restrict__ wkT, const float* __restrict__ bk,
    const bf16_t* __restrict__ wvT, const float* __restrict__ bv,
    const float* __restrict__ pos_bias, bf16_t* __restrict__ ctx)
{
  extern __shared__ char smem[];
  bf16_t* xs = (bf16_t*)smem;            // [33][1024]
  bf16_t* ks = xs + W_ * H_;             // [33][1024]
  bf16_t* vs = ks + W_ * H_;             // [33][1024]
  float*  qf = (float*)(vs + W_ * H_);   // [1024]
  float*  sc = qf + H_;                  // [16][40]

  int b = blockIdx.x, t = threadIdx.x;
  int lane = t & 31, wave = t >> 5;
  int l15 = lane & 15, lhi = (lane >> 4) * 16;

  // ---- phase 1: stage x = hidden[b] + pe as bf16 in LDS -------------------
  const float* xb = hidden + (size_t)b * (W_ * H_);
  for (int i = t * 4; i < W_ * H_; i += 256 * 4) {
    float4 hv = *(const float4*)(xb + i);
    float4 pv = *(const float4*)(pe + i);
    xs[i + 0] = f2bf(hv.x + pv.x);
    xs[i + 1] = f2bf(hv.y + pv.y);
    xs[i + 2] = f2bf(hv.z + pv.z);
    xs[i + 3] = f2bf(hv.w + pv.w);
  }
  __syncthreads();

  // ---- phase 2: q row (scalar f32; wq fits in L2) -------------------------
  {
    float qa[4];
#pragma unroll
    for (int i = 0; i < 4; ++i) qa[i] = bq[t + i * 256];
    for (int k = 0; k < H_; ++k) {
      float xk = bf2f(xs[MID_ * H_ + k]);
      const float* wr = wq + (size_t)k * H_;
#pragma unroll
      for (int i = 0; i < 4; ++i) qa[i] += xk * wr[t + i * 256];
    }
#pragma unroll
    for (int i = 0; i < 4; ++i) qf[t + i * 256] = qa[i];
  }

  // ---- phase 3: K and V projections via WMMA ------------------------------
  // 2048 output columns total (K: 0..1023 -> ks, V: 1024..2047 -> vs);
  // each wave owns 256 columns, processed as 8 chunks of 32 (2 N-tiles)
  for (int chunk = 0; chunk < 8; ++chunk) {
    int gcol = wave * 256 + chunk * 32;
    int isV  = gcol >= H_;
    int ncol = isV ? (gcol - H_) : gcol;
    const bf16_t* Wt   = isV ? wvT : wkT;
    bf16_t*       dst  = isV ? vs  : ks;
    const float*  bias = isV ? bv  : bk;

    v8f acc[3][2];
    v8f zero = {0.f, 0.f, 0.f, 0.f, 0.f, 0.f, 0.f, 0.f};
#pragma unroll
    for (int mt = 0; mt < 3; ++mt)
#pragma unroll
      for (int nt = 0; nt < 2; ++nt) acc[mt][nt] = zero;

    for (int k0 = 0; k0 < H_; k0 += 32) {
      int kofs = k0 + lhi;
      v16bf a[3];
#pragma unroll
      for (int mt = 0; mt < 3; ++mt) {
        int row = mt * 16 + l15;
        if (row > 32) row = 32;                  // pad rows replicate row 32
        a[mt] = *(const v16bf*)(xs + row * H_ + kofs);
      }
      v16bf bb[2];
#pragma unroll
      for (int nt = 0; nt < 2; ++nt) {
        int n = ncol + nt * 16 + l15;
        bb[nt] = *(const v16bf*)(Wt + (size_t)n * H_ + kofs);
      }
#pragma unroll
      for (int mt = 0; mt < 3; ++mt)
#pragma unroll
        for (int nt = 0; nt < 2; ++nt)
          acc[mt][nt] = __builtin_amdgcn_wmma_f32_16x16x32_bf16(
              false, a[mt], false, bb[nt], (short)0, acc[mt][nt], false, false);
    }

    int rhi = (lane >= 16) ? 8 : 0;
#pragma unroll
    for (int mt = 0; mt < 3; ++mt)
#pragma unroll
      for (int nt = 0; nt < 2; ++nt)
#pragma unroll
        for (int r = 0; r < 8; ++r) {
          int row = mt * 16 + r + rhi;
          if (row < W_) {
            int c = ncol + nt * 16 + l15;
            dst[row * H_ + c] = f2bf(acc[mt][nt][r] + bias[c]);
          }
        }
  }
  __syncthreads();

  // ---- phase 4: scores ----------------------------------------------------
  const float scale = 1.0f / (8.0f + 1e-6f);     // 1/(sqrt(64)+1e-6)
  for (int s = t; s < NH_ * W_; s += 256) {
    int h = s / W_, kk = s - h * W_;
    const bf16_t* kr = ks + kk * H_ + h * DH_;
    const float*  qr = qf + h * DH_;
    float d = 0.f;
#pragma unroll 8
    for (int j = 0; j < DH_; ++j) d += qr[j] * bf2f(kr[j]);
    float pb = pos_bias[h * W_ + kk];
    pb = fminf(fmaxf(pb, -10.f), 10.f);
    float sv = d * scale + pb;
    sv = fminf(fmaxf(sv, -10.f), 10.f);
    sc[h * 40 + kk] = sv;
  }
  __syncthreads();

  // ---- phase 5: softmax (one thread per head; 33 elements) ----------------
  if (t < NH_) {
    float mx = -1e30f;
    for (int k = 0; k < W_; ++k) mx = fmaxf(mx, sc[t * 40 + k]);
    float sum = 0.f;
    for (int k = 0; k < W_; ++k) {
      float e = __expf(sc[t * 40 + k] - mx);
      sc[t * 40 + k] = e;
      sum += e;
    }
    float invs = 1.0f / sum;
    for (int k = 0; k < W_; ++k) sc[t * 40 + k] *= invs;
  }
  __syncthreads();

  // ---- phase 6: context ---------------------------------------------------
  for (int j = t; j < H_; j += 256) {
    int h = j >> 6;
    float a = 0.f;
    for (int k = 0; k < W_; ++k) a += sc[h * 40 + k] * bf2f(vs[k * H_ + j]);
    ctx[(size_t)b * H_ + j] = f2bf(a);
  }
}

// ---------------------------------------------------------------------------
// Kernel 4: WMMA GEMM  C[M][N] = A[M][K](bf16) * Bt[N][K](bf16)^T + bias
//   block = 256 threads (8 waves, 2x4), block tile 128x128, k-step 32.
//   A/B tiles are staged into double-buffered LDS by the Tensor Data Mover
//   (tensor_load_to_lds, TENSORcnt) issued from wave 0; compute waves feed
//   WMMA from ds_load_b128 fragments.
// ---------------------------------------------------------------------------
__global__ __launch_bounds__(256) void gemm_bf16_kernel(
    const bf16_t* __restrict__ A, const bf16_t* __restrict__ Bt,
    const float* __restrict__ bias, float* __restrict__ C,
    int M, int N, int K)
{
  __shared__ bf16_t As[2][128 * 32];
  __shared__ bf16_t Bs[2][128 * 32];

  int t = threadIdx.x, lane = t & 31, wave = t >> 5;
  int mw = wave >> 2, nw = wave & 3;
  int m0 = blockIdx.y * 128, n0 = blockIdx.x * 128;
  int l15 = lane & 15, lhi = (lane >> 4) * 16;

  unsigned as_base = (unsigned)(size_t)(void*)&As[0][0];
  unsigned bs_base = (unsigned)(size_t)(void*)&Bs[0][0];
  const unsigned bufbytes = 128 * 32 * 2;       // 8 KB per buffer
  bool issuer = (wave == 0);                    // one wave drives the TDM

  v8f acc[4][2];
  v8f zero = {0.f, 0.f, 0.f, 0.f, 0.f, 0.f, 0.f, 0.f};
#pragma unroll
  for (int mt = 0; mt < 4; ++mt)
#pragma unroll
    for (int nt = 0; nt < 2; ++nt) acc[mt][nt] = zero;

  // prologue: DMA first A/B tile pair into buffer 0
  if (issuer) {
    tdm_load_2d_bf16(as_base, A + (size_t)m0 * K, K, M, 32, 128);
    tdm_load_2d_bf16(bs_base, Bt + (size_t)n0 * K, K, N, 32, 128);
  }

  for (int k0 = 0; k0 < K; k0 += 32) {
    int buf = (k0 >> 5) & 1;
    if (issuer) {
      if (k0 + 32 < K) {
        // DMA next tile pair into the other buffer, then wait for current
        unsigned nb = (unsigned)(buf ^ 1) * bufbytes;
        tdm_load_2d_bf16(as_base + nb, A + (size_t)m0 * K + k0 + 32, K, M, 32, 128);
        tdm_load_2d_bf16(bs_base + nb, Bt + (size_t)n0 * K + k0 + 32, K, N, 32, 128);
        __builtin_amdgcn_s_wait_tensorcnt(2);
      } else {
        __builtin_amdgcn_s_wait_tensorcnt(0);
      }
    }
    __syncthreads();

    v16bf a[4], bb[2];
#pragma unroll
    for (int mt = 0; mt < 4; ++mt)
      a[mt] = *(const v16bf*)(&As[buf][0] + (mw * 64 + mt * 16 + l15) * 32 + lhi);
#pragma unroll
    for (int nt = 0; nt < 2; ++nt)
      bb[nt] = *(const v16bf*)(&Bs[buf][0] + (nw * 32 + nt * 16 + l15) * 32 + lhi);
#pragma unroll
    for (int mt = 0; mt < 4; ++mt)
#pragma unroll
      for (int nt = 0; nt < 2; ++nt)
        acc[mt][nt] = __builtin_amdgcn_wmma_f32_16x16x32_bf16(
            false, a[mt], false, bb[nt], (short)0, acc[mt][nt], false, false);
    __syncthreads();
  }

  int rhi = (lane >= 16) ? 8 : 0;
#pragma unroll
  for (int mt = 0; mt < 4; ++mt)
#pragma unroll
    for (int nt = 0; nt < 2; ++nt) {
      int col = n0 + nw * 32 + nt * 16 + l15;
      float bv = bias[col];
#pragma unroll
      for (int r = 0; r < 8; ++r) {
        int row = m0 + mw * 64 + mt * 16 + r + rhi;
        C[(size_t)row * N + col] = acc[mt][nt][r] + bv;
      }
    }
}

// ---------------------------------------------------------------------------
// Kernel 5: row LayerNorm + exact GELU + bf16 convert (row cached in LDS)
// ---------------------------------------------------------------------------
__global__ __launch_bounds__(256) void ln_gelu_kernel(
    const float* __restrict__ X, const float* __restrict__ g,
    const float* __restrict__ be, bf16_t* __restrict__ Y, int N)
{
  __shared__ float red[256];
  __shared__ float xrow[4096];
  int rowi = blockIdx.x, t = threadIdx.x;
  const float* x = X + (size_t)rowi * N;
  float s = 0.f, s2 = 0.f;
  for (int j = t; j < N; j += 256) {
    float v = x[j];
    xrow[j] = v;
    s += v; s2 += v * v;
  }
  float mean = block_reduce_sum(s, red) / (float)N;
  float m2   = block_reduce_sum(s2, red) / (float)N;
  float inv  = rsqrtf(m2 - mean * mean + 1e-5f);
  bf16_t* y = Y + (size_t)rowi * N;
  for (int j = t; j < N; j += 256) {
    float v = (xrow[j] - mean) * inv * g[j] + be[j];
    float ge = 0.5f * v * (1.0f + erff(v * 0.70710678118654752f));
    y[j] = f2bf(ge);
  }
}

// ---------------------------------------------------------------------------
// Host side
// ---------------------------------------------------------------------------
extern "C" void kernel_launch(void* const* d_in, const int* in_sizes, int n_in,
                              void* d_out, int out_size, void* d_ws, size_t ws_size,
                              hipStream_t stream) {
  (void)in_sizes; (void)n_in; (void)out_size; (void)ws_size;

  const float* hidden   = (const float*)d_in[0];
  // d_in[1], d_in[2] (target_clip_idx, total_num_clips) unused by reference
  const float* wq    = (const float*)d_in[3];
  const float* bq    = (const float*)d_in[4];
  const float* wk    = (const float*)d_in[5];
  const float* bk    = (const float*)d_in[6];
  const float* wv    = (const float*)d_in[7];
  const float* bv    = (const float*)d_in[8];
  const float* pos_b = (const float*)d_in[9];
  const float* pe_w1 = (const float*)d_in[10];
  const float* pe_b1 = (const float*)d_in[11];
  const float* pe_g1 = (const float*)d_in[12];
  const float* pe_be1= (const float*)d_in[13];
  const float* pe_w2 = (const float*)d_in[14];
  const float* pe_b2 = (const float*)d_in[15];
  const float* pe_g2 = (const float*)d_in[16];
  const float* pe_be2= (const float*)d_in[17];
  const float* o_w1  = (const float*)d_in[18];
  const float* o_b1  = (const float*)d_in[19];
  const float* o_g1  = (const float*)d_in[20];
  const float* o_be1 = (const float*)d_in[21];
  const float* o_w2  = (const float*)d_in[22];
  const float* o_b2  = (const float*)d_in[23];
  const float* o_g2  = (const float*)d_in[24];
  const float* o_be2 = (const float*)d_in[25];
  const float* o_w3  = (const float*)d_in[26];
  const float* o_b3  = (const float*)d_in[27];
  const float* o_g3  = (const float*)d_in[28];
  const float* o_be3 = (const float*)d_in[29];
  const float* o_w4  = (const float*)d_in[30];
  const float* o_b4  = (const float*)d_in[31];

  char* ws = (char*)d_ws;
  size_t off = 0;
  auto take = [&](size_t bytes) -> char* {
    char* p = ws + off;
    off += (bytes + 255) & ~(size_t)255;
    return p;
  };

  float*  pe   = (float*) take((size_t)W_ * H_ * 4);
  bf16_t* wkT  = (bf16_t*)take((size_t)H_ * H_ * 2);
  bf16_t* wvT  = (bf16_t*)take((size_t)H_ * H_ * 2);
  bf16_t* w1T  = (bf16_t*)take((size_t)2048 * 1024 * 2);
  bf16_t* w2T  = (bf16_t*)take((size_t)4096 * 2048 * 2);
  bf16_t* w3T  = (bf16_t*)take((size_t)2048 * 4096 * 2);
  bf16_t* w4T  = (bf16_t*)take((size_t)1024 * 2048 * 2);
  bf16_t* ctx  = (bf16_t*)take((size_t)B_ * H_ * 2);
  float*  hbuf = (float*) take((size_t)B_ * 4096 * 4);
  bf16_t* a1   = (bf16_t*)take((size_t)B_ * 2048 * 2);
  bf16_t* a2   = (bf16_t*)take((size_t)B_ * 4096 * 2);
  bf16_t* a3   = (bf16_t*)take((size_t)B_ * 2048 * 2);

  // 1) positional encoding (once; batch independent)
  pe_kernel<<<W_, 256, 0, stream>>>(pe_w1, pe_b1, pe_g1, pe_be1,
                                    pe_w2, pe_b2, pe_g2, pe_be2, pe);

  // 2) weight conversion/transpose to bf16
  auto launchT = [&](const float* Wsrc, bf16_t* Wdst, int K, int N) {
    size_t total = (size_t)K * N;
    conv_transpose_kernel<<<(unsigned)((total + 255) / 256), 256, 0, stream>>>(
        Wsrc, Wdst, K, N);
  };
  launchT(wk,   wkT, 1024, 1024);
  launchT(wv,   wvT, 1024, 1024);
  launchT(o_w1, w1T, 1024, 2048);
  launchT(o_w2, w2T, 2048, 4096);
  launchT(o_w3, w3T, 4096, 2048);
  launchT(o_w4, w4T, 2048, 1024);

  // 3) fused per-batch attention (dynamic LDS ~205 KB; CDNA5 WGP has 320 KB)
  size_t attn_lds = (size_t)(3 * W_ * H_) * 2   // xs, ks, vs (bf16)
                  + (size_t)H_ * 4              // qf
                  + (size_t)NH_ * 40 * 4;       // sc
  attn_kernel<<<B_, 256, attn_lds, stream>>>(hidden, pe, wq, bq,
                                             wkT, bk, wvT, bv, pos_b, ctx);

  // 4) output MLP: 4 WMMA GEMMs (TDM-staged) with LN+GELU between
  gemm_bf16_kernel<<<dim3(2048 / 128, B_ / 128), 256, 0, stream>>>(
      ctx, w1T, o_b1, hbuf, B_, 2048, 1024);
  ln_gelu_kernel<<<B_, 256, 0, stream>>>(hbuf, o_g1, o_be1, a1, 2048);

  gemm_bf16_kernel<<<dim3(4096 / 128, B_ / 128), 256, 0, stream>>>(
      a1, w2T, o_b2, hbuf, B_, 4096, 2048);
  ln_gelu_kernel<<<B_, 256, 0, stream>>>(hbuf, o_g2, o_be2, a2, 4096);

  gemm_bf16_kernel<<<dim3(2048 / 128, B_ / 128), 256, 0, stream>>>(
      a2, w3T, o_b3, hbuf, B_, 2048, 4096);
  ln_gelu_kernel<<<B_, 256, 0, stream>>>(hbuf, o_g3, o_be3, a3, 2048);

  gemm_bf16_kernel<<<dim3(1024 / 128, B_ / 128), 256, 0, stream>>>(
      a3, w4T, o_b4, (float*)d_out, B_, 1024, 2048);
}
